// Attention_29910152249410
// MI455X (gfx1250) — matrix operand order repaired and checked by reading, not compiled, not run on hardware
//
#include <hip/hip_runtime.h>
#include <stdint.h>

#define HID   1024
#define BATCH 32
#define SRC   2048
#define NT    64        // N tiles  (1024 / 16)
#define KSTEPS 32       // K steps  (1024 / 32)
#define ROW_BYTES (HID * 2 + 16)   // padded bf16 row in LDS (2064 B)

typedef __bf16 v16bf __attribute__((ext_vector_type(16)));
typedef float  v8f   __attribute__((ext_vector_type(8)));

struct Bits256 { uint4 lo, hi; };

__device__ __forceinline__ uint32_t pack2_bf16(float a, float b) {
  // round-to-nearest-even fp32 -> bf16, packed pair
  uint32_t ua = __float_as_uint(a);
  uint32_t ub = __float_as_uint(b);
  ua = (ua + 0x7FFFu + ((ua >> 16) & 1u)) >> 16;
  ub = (ub + 0x7FFFu + ((ub >> 16) & 1u)) >> 16;
  return ua | (ub << 16);
}

// Branch-free tanh: prefer the CDNA5 hardware transcendental (v_tanh_f32),
// fall back to an exp-based identity (saturates correctly, no exec divergence).
__device__ __forceinline__ float fast_tanh(float x) {
#if __has_builtin(__builtin_amdgcn_tanhf)
  return __builtin_amdgcn_tanhf(x);
#elif __has_builtin(__builtin_amdgcn_tanh_f32)
  return __builtin_amdgcn_tanh_f32(x);
#else
  float e = __expf(2.0f * x);          // v_mul + v_exp_f32
  return 1.0f - 2.0f / (e + 1.0f);     // v_add + v_rcp_f32 + v_fma
#endif
}

// ---------------------------------------------------------------------------
// Prep A: hb[b,o] = dot(hidden[b,:], w_h[o,:]) + bias[o]     (32x1024 fp32)
// ---------------------------------------------------------------------------
__global__ void k_hproj(const float* __restrict__ hidden,
                        const float* __restrict__ attn_w,
                        const float* __restrict__ attn_b,
                        float* __restrict__ hb) {
  int id = blockIdx.x * blockDim.x + threadIdx.x;   // id = b*1024 + o
  int b = id >> 10, o = id & (HID - 1);
  const float4* hp = (const float4*)(hidden + (size_t)b * HID);
  const float4* wp = (const float4*)(attn_w + (size_t)o * (2 * HID)); // w_h row o
  float acc = 0.0f;
  for (int k = 0; k < HID / 4; ++k) {
    float4 h = hp[k], w = wp[k];
    acc += h.x * w.x + h.y * w.y + h.z * w.z + h.w * w.w;
  }
  hb[id] = acc + attn_b[o];
}

// ---------------------------------------------------------------------------
// Prep B: pre-swizzle w_e into bf16 WMMA B-fragment layout.
// Fragment (t, ks, lane) holds 16 bf16: w_e[t*16 + lane%16][ks*32 + 16*(lane/16) + 0..15]
// Stored contiguously: element-group id = (t*32 + ks)*32 + lane, 32 bytes each.
// ---------------------------------------------------------------------------
__global__ void k_bswz(const float* __restrict__ attn_w, uint4* __restrict__ bbuf) {
  int id = blockIdx.x * blockDim.x + threadIdx.x;   // 0 .. 65535
  int L  = id & 31;
  int ks = (id >> 5) & 31;
  int t  = id >> 10;
  int n  = t * 16 + (L & 15);
  int kb = ks * 32 + ((L >> 4) << 4);
  const float4* src = (const float4*)(attn_w + (size_t)n * (2 * HID) + HID + kb);
  float4 f0 = src[0], f1 = src[1], f2 = src[2], f3 = src[3];
  uint4 o0, o1;
  o0.x = pack2_bf16(f0.x, f0.y); o0.y = pack2_bf16(f0.z, f0.w);
  o0.z = pack2_bf16(f1.x, f1.y); o0.w = pack2_bf16(f1.z, f1.w);
  o1.x = pack2_bf16(f2.x, f2.y); o1.y = pack2_bf16(f2.z, f2.w);
  o1.z = pack2_bf16(f3.x, f3.y); o1.w = pack2_bf16(f3.z, f3.w);
  bbuf[(size_t)id * 2]     = o0;
  bbuf[(size_t)id * 2 + 1] = o1;
}

// ---------------------------------------------------------------------------
// Main fused kernel: per block = 16 rows of (b,s).  8 waves, each owns 8 N-tiles.
// scores[row] = sum_o tanh(e_proj[row,o] + hb[b,o]) * v[o]
// ---------------------------------------------------------------------------
__global__ __launch_bounds__(256) void k_energy(
    const float* __restrict__ enc,
    const uint4* __restrict__ bbuf,
    const float* __restrict__ hb,
    const float* __restrict__ v,
    float* __restrict__ scores) {
  __shared__ __align__(16) unsigned char atile[16 * ROW_BYTES];
  __shared__ float ssum[16];

  const int tid  = threadIdx.x;
  const int lane = tid & 31;
  const int wave = tid >> 5;
  const int row_base = blockIdx.x * 16;
  const int b = row_base >> 11;   // 2048 rows per batch; tile never crosses batch

  // --- stage 16 x 1024 A-panel (fp32 -> bf16) into padded LDS ---
  for (int it = 0; it < 8; ++it) {
    int g   = it * 256 + tid;     // 0..2047 groups of 8 elements
    int row = g >> 7;
    int kg  = g & 127;
    const float4* ep = (const float4*)(enc + (size_t)(row_base + row) * HID + kg * 8);
    float4 f0 = ep[0], f1 = ep[1];
    uint4 o;
    o.x = pack2_bf16(f0.x, f0.y);
    o.y = pack2_bf16(f0.z, f0.w);
    o.z = pack2_bf16(f1.x, f1.y);
    o.w = pack2_bf16(f1.z, f1.w);
    *(uint4*)(atile + row * ROW_BYTES + kg * 16) = o;
  }
  if (tid < 16) ssum[tid] = 0.0f;
  __syncthreads();

  // A-fragment addressing (contiguous K, matching B fragments):
  // lanes 0-15:  row = lane,     K = k0+0..15
  // lanes 16-31: row = lane-16,  K = k0+16..31
  const int arow = lane & 15;
  const int asel = (lane >> 4) << 5;   // +32 bytes for upper half

  float part[8];
#pragma unroll
  for (int r = 0; r < 8; ++r) part[r] = 0.0f;

#pragma unroll 1
  for (int i = 0; i < 8; ++i) {
    const int t = wave * 8 + i;                       // N-tile index, n0 = t*16
    const float hbv = hb[b * HID + t * 16 + (lane & 15)];
    const float vv  = v[t * 16 + (lane & 15)];

    v8f c = {};
    const uint4* bp = bbuf + ((size_t)t * 1024 + lane) * 2;
    const unsigned char* abase = atile + arow * ROW_BYTES + asel;

    // Unroll 8: deep enough load pipeline ahead of each WMMA, while keeping
    // live fragments bounded so VGPR count stays < 256 (no MSB addressing).
#pragma unroll 8
    for (int ks = 0; ks < KSTEPS; ++ks) {
      Bits256 ab, bb;
      const unsigned char* ap = abase + ks * 64;      // k0 = ks*32 -> 64 bytes
      ab.lo = *(const uint4*)ap;
      ab.hi = *(const uint4*)(ap + 16);
      bb.lo = bp[0];
      bb.hi = bp[1];
      bp += 64;                                       // 32 lanes * 2 uint4
      v16bf af = __builtin_bit_cast(v16bf, ab);
      v16bf bf = __builtin_bit_cast(v16bf, bb);
      c = __builtin_amdgcn_wmma_f32_16x16x32_bf16(
          /*neg_a=*/false, af, /*neg_b=*/false, bf,
          /*c_mod=*/(short)0, c, /*reuse_a=*/false, /*reuse_b=*/false);
    }

    // C layout: VGPR r, lane L -> M = r + 8*(L/16), N = L%16
#pragma unroll
    for (int r = 0; r < 8; ++r) {
      float e = fast_tanh(c[r] + hbv);
      part[r] += e * vv;
    }
  }

  // reduce over N (16 lanes per half; xor<16 never crosses the half boundary)
#pragma unroll
  for (int off = 1; off < 16; off <<= 1) {
#pragma unroll
    for (int r = 0; r < 8; ++r)
      part[r] += __shfl_xor(part[r], off, 32);
  }
  if ((lane & 15) == 0) {
    int mbase = (lane >> 4) * 8;   // lane 0 -> rows 0..7, lane 16 -> rows 8..15
#pragma unroll
    for (int r = 0; r < 8; ++r)
      atomicAdd(&ssum[mbase + r], part[r]);
  }
  __syncthreads();
  if (tid < 16) scores[row_base + tid] = ssum[tid];
}

// ---------------------------------------------------------------------------
// Softmax over s (2048) per batch row
// ---------------------------------------------------------------------------
__global__ void k_softmax(const float* __restrict__ scores, float* __restrict__ out) {
  __shared__ float red[256];
  const int b = blockIdx.x;
  const int tid = threadIdx.x;
  const float* row = scores + (size_t)b * SRC;

  float m = -3.4e38f;
  for (int s = tid; s < SRC; s += 256) m = fmaxf(m, row[s]);
  red[tid] = m; __syncthreads();
  for (int o = 128; o > 0; o >>= 1) {
    if (tid < o) red[tid] = fmaxf(red[tid], red[tid + o]);
    __syncthreads();
  }
  m = red[0]; __syncthreads();

  float sum = 0.0f;
  for (int s = tid; s < SRC; s += 256) sum += __expf(row[s] - m);
  red[tid] = sum; __syncthreads();
  for (int o = 128; o > 0; o >>= 1) {
    if (tid < o) red[tid] += red[tid + o];
    __syncthreads();
  }
  float inv = 1.0f / red[0];
  for (int s = tid; s < SRC; s += 256)
    out[(size_t)b * SRC + s] = __expf(row[s] - m) * inv;
}

// ---------------------------------------------------------------------------
extern "C" void kernel_launch(void* const* d_in, const int* in_sizes, int n_in,
                              void* d_out, int out_size, void* d_ws, size_t ws_size,
                              hipStream_t stream) {
  const float* hidden = (const float*)d_in[0];
  const float* enc    = (const float*)d_in[1];
  const float* attn_w = (const float*)d_in[2];
  const float* attn_b = (const float*)d_in[3];
  const float* v      = (const float*)d_in[4];
  float* out = (float*)d_out;

  char*  ws     = (char*)d_ws;
  float* hb     = (float*)ws;                              // 32*1024 f32  = 128 KB
  uint4* bbuf   = (uint4*)(ws + 131072);                   // swizzled bf16 w_e = 2 MB
  float* scores = (float*)(ws + 131072 + 2097152);         // 65536 f32    = 256 KB

  k_hproj  <<<(BATCH * HID) / 256, 256, 0, stream>>>(hidden, attn_w, attn_b, hb);
  k_bswz   <<<(NT * KSTEPS * 32) / 256, 256, 0, stream>>>(attn_w, bbuf);
  k_energy <<<(BATCH * SRC) / 16, 256, 0, stream>>>(enc, bbuf, hb, v, scores);
  k_softmax<<<BATCH, 256, 0, stream>>>(scores, out);
}
